// RLSAttn_34325378629691
// MI455X (gfx1250) — compile-verified, matrix operand order checked by reading, not run on hardware
//
#include <hip/hip_runtime.h>
#include <hip/hip_bf16.h>

// RLS attention for MI455X (gfx1250, wave32, WMMA bf16 16x16x32).
// B=4 H=16 N=4096 D=64 SPLIT=64 G=64.

#define BB 4
#define HH 16
#define NN 4096
#define DD 64
#define GG 64            // N / SPLIT
#define SPLIT 64
#define EE (SPLIT * DD)  // 4096 flattened block width

typedef __attribute__((ext_vector_type(16))) __bf16 v16bf;
typedef __attribute__((ext_vector_type(2)))  __bf16 v2bf;
typedef __attribute__((ext_vector_type(8)))  float  v8f;

// ---- packed f32x2 -> bf16x2 ----
#if __has_builtin(__builtin_amdgcn_cvt_pk_bf16_f32)
__device__ __forceinline__ unsigned packbf2(float lo, float hi) {
    union { v2bf v; unsigned u; } c;
    c.v = __builtin_amdgcn_cvt_pk_bf16_f32(lo, hi);
    return c.u;
}
#else
// round-to-nearest (ties away) + byte-perm pack: 3 VALU per pair
__device__ __forceinline__ unsigned packbf2(float lo, float hi) {
    union { float f; unsigned u; } a, b; a.f = lo; b.f = hi;
    return __builtin_amdgcn_perm(b.u + 0x8000u, a.u + 0x8000u, 0x07060302u);
}
#endif

__device__ __forceinline__ v8f wmma_bf16(v16bf a, v16bf b, v8f c) {
    // (neg_a, A, neg_b, B, c_mod, C, reuse_a, reuse_b)
    return __builtin_amdgcn_wmma_f32_16x16x32_bf16(false, a, false, b,
                                                   (short)0, c, false, false);
}

union FragU { v16bf v; unsigned u[8]; uint4 q[2]; };

// A-fragment (16x32, lane m = rows) or B-fragment (32x16, lane n = product
// columns = ROWS of a row-major source, i.e. computes X @ Src^T).
// Lane L: chunk base kb = 8*(L>>4); elements [kb..kb+7] and [kb+16..kb+23].
__device__ __forceinline__ v16bf load_frag_rows_f32(const float* base, int stride,
                                                    int lane) {
    const float* p = base + (lane & 15) * stride + ((lane >> 4) << 3);
    float4 c0 = *(const float4*)(p);
    float4 c1 = *(const float4*)(p + 4);
    float4 c2 = *(const float4*)(p + 16);
    float4 c3 = *(const float4*)(p + 20);
    FragU f;
    f.u[0] = packbf2(c0.x, c0.y);
    f.u[1] = packbf2(c0.z, c0.w);
    f.u[2] = packbf2(c1.x, c1.y);
    f.u[3] = packbf2(c1.z, c1.w);
    f.u[4] = packbf2(c2.x, c2.y);
    f.u[5] = packbf2(c2.z, c2.w);
    f.u[6] = packbf2(c3.x, c3.y);
    f.u[7] = packbf2(c3.z, c3.w);
    return f.v;
}

// bf16 source with 16B-aligned rows: two b128 loads.
__device__ __forceinline__ v16bf load_frag_rows_bf16(const __bf16* base, int stride,
                                                     int lane) {
    const __bf16* p = base + (lane & 15) * stride + ((lane >> 4) << 3);
    FragU f;
    f.q[0] = *(const uint4*)(p);
    f.q[1] = *(const uint4*)(p + 16);
    return f.v;
}

// ---------------- Stage 1: per-(b,h,g) local attention ----------------
// grid = B*H*G blocks, 128 threads (4 waves). Wave w owns output rows [16w,16w+16).
__global__ __launch_bounds__(128) void rls_stage1(const float* __restrict__ Q,
                                                  const float* __restrict__ K,
                                                  const float* __restrict__ V,
                                                  float* __restrict__ x1) {
    const int blk = blockIdx.x;            // b*h*g
    const int g   = blk & (GG - 1);
    const int bh  = blk >> 6;
    const size_t tile = ((size_t)bh * NN + (size_t)g * SPLIT) * DD;
    const float* Qt = Q + tile;
    const float* Kt = K + tile;
    const float* Vt = V + tile;
    float*       Xt = x1 + tile;

    __shared__ float  S[SPLIT * SPLIT];    // 16 KB logits
    __shared__ __bf16 P[SPLIT * SPLIT];    // 8 KB probs
    __shared__ __bf16 KB[SPLIT * DD];      // 8 KB K tile (bf16, converted once)
    __shared__ __bf16 VT[DD * 72];         // 9 KB V^T, padded rows (stride 72)

    const int tid  = threadIdx.x;
    const int lane = tid & 31;
    const int wave = tid >> 5;
    const v8f vzero = {0.f, 0.f, 0.f, 0.f, 0.f, 0.f, 0.f, 0.f};

    // stage K (row-major, packed pairs)
    for (int i = tid; i < SPLIT * DD / 4; i += 128) {
        int row = i >> 4;                  // 16 float4-groups per 64-wide row
        int k4  = (i & 15) << 2;
        float4 c = *(const float4*)(Kt + row * DD + k4);
        *(uint2*)(KB + row * DD + k4) = make_uint2(packbf2(c.x, c.y), packbf2(c.z, c.w));
    }
    // stage V transposed: VT[d*72 + t]; two t per thread -> packed b32 store
    for (int i = tid; i < SPLIT * DD / 2; i += 128) {
        int d  = i & 63;
        int t2 = (i >> 6) << 1;
        float v0 = Vt[(size_t)t2 * DD + d];
        float v1 = Vt[(size_t)(t2 + 1) * DD + d];
        *(unsigned*)(VT + d * 72 + t2) = packbf2(v0, v1);
    }
    __syncthreads();

    // ---- S = Q @ K^T : 4 col-tiles x 2 k-steps (scale applied in softmax) ----
    v8f acc[4] = {vzero, vzero, vzero, vzero};
#pragma unroll
    for (int ks = 0; ks < 2; ++ks) {
        v16bf a = load_frag_rows_f32(Qt + (wave * 16) * DD + ks * 32, DD, lane);
#pragma unroll
        for (int ct = 0; ct < 4; ++ct) {
            v16bf b = load_frag_rows_bf16(KB + (ct * 16) * DD + ks * 32, DD, lane);
            acc[ct] = wmma_bf16(a, b, acc[ct]);
        }
    }
    {   // scatter C/D fragment (lane: col n = L%16, rows r + 8*(L/16))
        const int n = lane & 15, mo = (lane >> 4) << 3;
#pragma unroll
        for (int ct = 0; ct < 4; ++ct)
#pragma unroll
            for (int r = 0; r < 8; ++r)
                S[(wave * 16 + mo + r) * SPLIT + ct * 16 + n] = acc[ct][r];
    }
    __syncthreads();

    // ---- row softmax (with scale folded into exp) -> P (bf16, packed) ----
    if (tid < SPLIT) {
        const float scale1 = 0.125f;       // 64^-0.5
        float4* r4 = (float4*)(S + tid * SPLIT);
        float m = -3.402823466e38f;
#pragma unroll
        for (int j = 0; j < 16; ++j) {
            float4 v = r4[j];
            m = fmaxf(m, fmaxf(fmaxf(v.x, v.y), fmaxf(v.z, v.w)));
        }
        float s = 0.f;
#pragma unroll
        for (int j = 0; j < 16; ++j) {
            float4 v = r4[j];
            v.x = __expf((v.x - m) * scale1); v.y = __expf((v.y - m) * scale1);
            v.z = __expf((v.z - m) * scale1); v.w = __expf((v.w - m) * scale1);
            s += (v.x + v.y) + (v.z + v.w);
            r4[j] = v;
        }
        float inv = 1.f / s;
        unsigned* prow = (unsigned*)(P + tid * SPLIT);
#pragma unroll
        for (int j = 0; j < 16; ++j) {
            float4 v = r4[j];
            prow[2 * j]     = packbf2(v.x * inv, v.y * inv);
            prow[2 * j + 1] = packbf2(v.z * inv, v.w * inv);
        }
    }
    __syncthreads();

    // ---- X1 = P @ V (B from VT rows, stride 72) ----
    v8f xacc[4] = {vzero, vzero, vzero, vzero};
#pragma unroll
    for (int ks = 0; ks < 2; ++ks) {
        v16bf a = load_frag_rows_bf16(P + (wave * 16) * SPLIT + ks * 32, SPLIT, lane);
#pragma unroll
        for (int ct = 0; ct < 4; ++ct) {
            v16bf b = load_frag_rows_bf16(VT + (ct * 16) * 72 + ks * 32, 72, lane);
            xacc[ct] = wmma_bf16(a, b, xacc[ct]);
        }
    }
    {
        const int n = lane & 15, mo = (lane >> 4) << 3;
#pragma unroll
        for (int ct = 0; ct < 4; ++ct)
#pragma unroll
            for (int r = 0; r < 8; ++r)
                Xt[(wave * 16 + mo + r) * DD + ct * 16 + n] = xacc[ct][r];
    }
}

// ---------------- Stage 2a: coarse logits + softmax -> attn2 ----------------
// grid = B*H blocks. K-dim = 4096, processed in 16 chunks of 256; each chunk of
// k2 is converted to bf16 once into LDS and shared by all 4 waves.
__global__ __launch_bounds__(128) void rls_stage2_logits(const float* __restrict__ Q,
                                                         const float* __restrict__ K,
                                                         float* __restrict__ attn2) {
    const int bh = blockIdx.x;
    const float* q2 = Q + (size_t)bh * NN * DD;   // 64 x 4096
    const float* k2 = K + (size_t)bh * NN * DD;

    __shared__ float  S2[GG * GG];                // 16 KB
    __shared__ __bf16 KC[GG * 256];               // 32 KB: k2 chunk (64 x 256)

    const int tid  = threadIdx.x;
    const int lane = tid & 31;
    const int wave = tid >> 5;
    const v8f vzero = {0.f, 0.f, 0.f, 0.f, 0.f, 0.f, 0.f, 0.f};

    v8f acc[4] = {vzero, vzero, vzero, vzero};
    const float* abase = q2 + (wave * 16) * EE;

    for (int chunk = 0; chunk < 16; ++chunk) {
        const float* kcg = k2 + chunk * 256;
        for (int i = tid; i < GG * 256 / 4; i += 128) {   // coalesced float4 stage
            int row = i >> 6;                              // 64 groups per row
            int k4  = (i & 63) << 2;
            float4 c = *(const float4*)(kcg + (size_t)row * EE + k4);
            *(uint2*)(KC + row * 256 + k4) =
                make_uint2(packbf2(c.x, c.y), packbf2(c.z, c.w));
        }
        __syncthreads();
#pragma unroll
        for (int ks = 0; ks < 8; ++ks) {
            __builtin_prefetch(abase + chunk * 256 + ks * 32 + 512, 0, 1);
            v16bf a = load_frag_rows_f32(abase + chunk * 256 + ks * 32, EE, lane);
#pragma unroll
            for (int ct = 0; ct < 4; ++ct) {
                v16bf b = load_frag_rows_bf16(KC + (ct * 16) * 256 + ks * 32, 256, lane);
                acc[ct] = wmma_bf16(a, b, acc[ct]);
            }
        }
        __syncthreads();
    }
    {
        const int n = lane & 15, mo = (lane >> 4) << 3;
#pragma unroll
        for (int ct = 0; ct < 4; ++ct)
#pragma unroll
            for (int r = 0; r < 8; ++r)
                S2[(wave * 16 + mo + r) * GG + ct * 16 + n] = acc[ct][r];
    }
    __syncthreads();

    if (tid < GG) {
        const float scale2 = 0.015625f;           // (64*64)^-0.5 = 1/64
        float4* r4 = (float4*)(S2 + tid * GG);
        float m = -3.402823466e38f;
#pragma unroll
        for (int j = 0; j < 16; ++j) {
            float4 v = r4[j];
            m = fmaxf(m, fmaxf(fmaxf(v.x, v.y), fmaxf(v.z, v.w)));
        }
        float s = 0.f;
#pragma unroll
        for (int j = 0; j < 16; ++j) {
            float4 v = r4[j];
            v.x = __expf((v.x - m) * scale2); v.y = __expf((v.y - m) * scale2);
            v.z = __expf((v.z - m) * scale2); v.w = __expf((v.w - m) * scale2);
            s += (v.x + v.y) + (v.z + v.w);
            r4[j] = v;
        }
        float inv = 1.f / s;
        float4* arow = (float4*)(attn2 + (size_t)bh * GG * GG + tid * GG);
#pragma unroll
        for (int j = 0; j < 16; ++j) {
            float4 v = r4[j];
            v.x *= inv; v.y *= inv; v.z *= inv; v.w *= inv;
            arow[j] = v;
        }
    }
}

// ---------------- Stage 2b: out = attn2 @ x1_grid ----------------
// grid = (16 e-strips of 256, B*H). x1 strip staged TRANSPOSED in LDS (bf16,
// padded stride 72) so B fragments are two b128 LDS loads per lane.
__global__ __launch_bounds__(128) void rls_stage2_out(const float* __restrict__ attn2,
                                                      const float* __restrict__ x1,
                                                      float* __restrict__ out) {
    const int strip = blockIdx.x;                 // 0..15 (256 columns each)
    const int bh    = blockIdx.y;

    __shared__ __bf16 XST[256 * 72];              // 36 KB: XST[ec*72 + f]

    const int tid  = threadIdx.x;
    const int lane = tid & 31;
    const int wave = tid >> 5;
    const v8f vzero = {0.f, 0.f, 0.f, 0.f, 0.f, 0.f, 0.f, 0.f};

    // transpose-stage: two f per thread (coalesced global reads, packed b32 store)
    const float* xp = x1 + (size_t)bh * GG * EE + (size_t)strip * 256;
    for (int i = tid; i < GG * 256 / 2; i += 128) {
        int ec = i & 255;
        int f2 = (i >> 8) << 1;
        float v0 = xp[(size_t)f2 * EE + ec];
        float v1 = xp[(size_t)(f2 + 1) * EE + ec];
        *(unsigned*)(XST + ec * 72 + f2) = packbf2(v0, v1);
    }
    __syncthreads();

    // preload both A fragments (K = g = 64 -> 2 k-steps)
    const float* a2 = attn2 + (size_t)bh * GG * GG;
    v16bf afr[2];
#pragma unroll
    for (int ks = 0; ks < 2; ++ks)
        afr[ks] = load_frag_rows_f32(a2 + (wave * 16) * GG + ks * 32, GG, lane);

    float* op = out + (size_t)bh * GG * EE + (size_t)strip * 256;
    const int n = lane & 15, mo = (lane >> 4) << 3;
#pragma unroll
    for (int ct = 0; ct < 16; ++ct) {
        v8f acc = vzero;
#pragma unroll
        for (int ks = 0; ks < 2; ++ks) {
            v16bf b = load_frag_rows_bf16(XST + (ct * 16) * 72 + ks * 32, 72, lane);
            acc = wmma_bf16(afr[ks], b, acc);
        }
#pragma unroll
        for (int r = 0; r < 8; ++r)
            op[(size_t)(wave * 16 + mo + r) * EE + ct * 16 + n] = acc[r];
    }
}

extern "C" void kernel_launch(void* const* d_in, const int* in_sizes, int n_in,
                              void* d_out, int out_size, void* d_ws, size_t ws_size,
                              hipStream_t stream) {
    const float* Q = (const float*)d_in[0];
    const float* K = (const float*)d_in[1];
    const float* V = (const float*)d_in[2];
    float* out = (float*)d_out;

    // workspace: x1 (B*H*N*D f32 = 64 MB) | attn2 (B*H*G*G f32 = 1 MB)
    float* x1    = (float*)d_ws;
    float* attn2 = x1 + (size_t)BB * HH * NN * DD;

    rls_stage1<<<BB * HH * GG, 128, 0, stream>>>(Q, K, V, x1);
    rls_stage2_logits<<<BB * HH, 128, 0, stream>>>(Q, K, attn2);
    dim3 g3(EE / 256, BB * HH);
    rls_stage2_out<<<g3, 128, 0, stream>>>(attn2, x1, out);
}
